// ResidualAttentionBlockMoE_45268955300333
// MI455X (gfx1250) — matrix operand order repaired and checked by reading, not compiled, not run on hardware
//
#include <hip/hip_runtime.h>

#define N_    211
#define B_    32
#define D_    768
#define H_    12
#define HD_   64
#define E_    8
#define DF_   3072
#define T_    (N_ * B_)        // 6752 tokens, 6752/16 = 422 exact
#define QKVLD 2304
#define KCH   128              // K-chunk staged in LDS (bf16 elements)
#define QT_   14               // ceil(211/16)
#define SC_W  (QT_ * 16)       // 224 padded score width

typedef __bf16 bf16;
typedef __attribute__((ext_vector_type(16))) __bf16 v16bf;
typedef __attribute__((ext_vector_type(8)))  __bf16 v8bf;
typedef __attribute__((ext_vector_type(8)))  float  v8f;
typedef __attribute__((ext_vector_type(4)))  unsigned int v4u;
typedef __attribute__((ext_vector_type(8)))  int v8i;
typedef __attribute__((ext_vector_type(4)))  int v4i;

#ifndef __has_builtin
#define __has_builtin(x) 0
#endif
#if __has_builtin(__builtin_amdgcn_tensor_load_to_lds)
#define HAVE_TDM 1
#else
#define HAVE_TDM 0
#endif

// ---------------- WMMA fragment helpers (layouts per CDNA5 ISA) ----

__device__ __forceinline__ v16bf zero16() {
  v16bf z;
#pragma unroll
  for (int i = 0; i < 16; ++i) z[i] = (bf16)0.0f;
  return z;
}

// A fragment (16x32 bf16): lane holds row (lane&15); elements 0..7 at K=k0+8*hi,
// elements 8..15 at K=k0+16+8*hi. Caller passes p = &A[row][k0 + 8*hi].
__device__ __forceinline__ v16bf ld_a_pair(const bf16* p) {
  v8bf lo = *(const v8bf*)p;
  v8bf h8 = *(const v8bf*)(p + 16);
  v16bf a;
#pragma unroll
  for (int i = 0; i < 8; ++i) { a[i] = lo[i]; a[i + 8] = h8[i]; }
  return a;
}

// B fragment (32x16 bf16): lane holds column (lane&15); elements 0..7 at
// K = k0 + 8*hi + e, elements 8..15 at K = k0 + 16 + 8*hi + (e-8). p = &W[col][k0 + 8*hi].
__device__ __forceinline__ v16bf ld_b_cont(const bf16* p) {
  v8bf lo = *(const v8bf*)p;
  v8bf h8 = *(const v8bf*)(p + 16);
  v16bf b;
#pragma unroll
  for (int i = 0; i < 8; ++i) { b[i] = lo[i]; b[i + 8] = h8[i]; }
  return b;
}

__device__ __forceinline__ v8f wmma_bf16(v16bf a, v16bf b, v8f c) {
  return __builtin_amdgcn_wmma_f32_16x16x32_bf16(false, a, false, b, (short)0, c,
                                                 false, false);
}

// ---- Tensor Data Mover: async 2-D tile load Global->LDS (D# per CDNA5 ISA)
// Dense row-major tile: tile_d0 x tile_d1 (bf16 elements), row stride = stride0.
__device__ __forceinline__ void tdm_load_a_tile(const bf16* gtile, void* lds,
                                                unsigned tensor_d0, unsigned tensor_d1,
                                                unsigned tile_d0, unsigned tile_d1,
                                                unsigned stride0) {
#if HAVE_TDM
  unsigned lds_off = (unsigned)(unsigned long long)(size_t)lds;
  unsigned long long ga = (unsigned long long)(size_t)gtile;
  v4u g0;
  g0[0] = 1u;                                             // count=1, user mode
  g0[1] = lds_off;                                        // lds_addr (bytes)
  g0[2] = (unsigned)(ga & 0xffffffffull);                 // global_addr[31:0]
  g0[3] = (unsigned)((ga >> 32) & 0x01ffffffull) | (2u << 30);  // addr[56:32] | type=2
  v8i g1;
  g1[0] = (int)(1u << 16);                                // data_size=1 (2B), wg_mask=0
  g1[1] = (int)((tensor_d0 & 0xffffu) << 16);             // tensor_dim0[15:0]
  g1[2] = (int)((tensor_d0 >> 16) | ((tensor_d1 & 0xffffu) << 16));
  g1[3] = (int)((tensor_d1 >> 16) | (tile_d0 << 16));     // tile_dim0
  g1[4] = (int)(tile_d1 & 0xffffu);                       // tile_dim1, tile_dim2=0
  g1[5] = (int)stride0;                                   // tensor_dim0_stride[31:0]
  g1[6] = 0;
  g1[7] = 0;
  v4i z = {0, 0, 0, 0};
#if __clang_major__ >= 23
  v8i z8 = {0, 0, 0, 0, 0, 0, 0, 0};
  __builtin_amdgcn_tensor_load_to_lds(g0, g1, z, z, z8, 0);
#else
  __builtin_amdgcn_tensor_load_to_lds(g0, g1, z, z, 0);
#endif
#if __has_builtin(__builtin_amdgcn_s_wait_tensorcnt)
  __builtin_amdgcn_s_wait_tensorcnt(0);
#endif
#else
  (void)gtile; (void)lds; (void)tensor_d0; (void)tensor_d1;
  (void)tile_d0; (void)tile_d1; (void)stride0;
#endif
}

// ---------------- elementwise / utility kernels ------------------------------

__global__ void cvt_f32_bf16(const float* __restrict__ in, bf16* __restrict__ out, int n) {
  int i = blockIdx.x * blockDim.x + threadIdx.x;
  if (i < n) out[i] = (bf16)in[i];
}

__global__ void zero_f32(float* p, int n) {
  int i = blockIdx.x * blockDim.x + threadIdx.x;
  if (i < n) p[i] = 0.0f;
}

__global__ void zero_i32(int* p, int n) {
  int i = blockIdx.x * blockDim.x + threadIdx.x;
  if (i < n) p[i] = 0;
}

// LayerNorm over D_=768, input fp32 row r, output bf16 row (remap? b*N+n : r)
__global__ void ln_kernel(const float* __restrict__ X, const float* __restrict__ g,
                          const float* __restrict__ beta, bf16* __restrict__ out,
                          int remap) {
  __shared__ float red[256];
  int r = blockIdx.x;
  const float* xr = X + (size_t)r * D_;
  float s = 0.0f;
  for (int i = threadIdx.x; i < D_; i += 256) s += xr[i];
  red[threadIdx.x] = s;
  __syncthreads();
  for (int st = 128; st > 0; st >>= 1) {
    if (threadIdx.x < st) red[threadIdx.x] += red[threadIdx.x + st];
    __syncthreads();
  }
  float mean = red[0] * (1.0f / D_);
  __syncthreads();
  float v = 0.0f;
  for (int i = threadIdx.x; i < D_; i += 256) { float d = xr[i] - mean; v += d * d; }
  red[threadIdx.x] = v;
  __syncthreads();
  for (int st = 128; st > 0; st >>= 1) {
    if (threadIdx.x < st) red[threadIdx.x] += red[threadIdx.x + st];
    __syncthreads();
  }
  float rstd = rsqrtf(red[0] * (1.0f / D_) + 1e-5f);
  int orow = remap ? ((r % B_) * N_ + (r / B_)) : r;
  bf16* op = out + (size_t)orow * D_;
  for (int i = threadIdx.x; i < D_; i += 256)
    op[i] = (bf16)((xr[i] - mean) * rstd * g[i] + beta[i]);
}

// ---------------- WMMA GEMMs -------------------------------------------------
// C[M,Nn] = A[M,Kd] @ W[Nn,Kd]^T (+bias). 8 waves/block; each wave owns a 16x64
// tile (4 accumulators) so the A fragment is reused 4x. 16-row A tile staged in
// LDS via TDM (tensor_load_to_lds) when available, else cooperative loads.

__global__ void gemm_bias_bf16out(const bf16* __restrict__ A, const bf16* __restrict__ W,
                                  const float* __restrict__ bias, bf16* __restrict__ C,
                                  int Nn, int Kd) {
  __shared__ bf16 sA[16 * KCH];
  int mt = blockIdx.y;
  int wave = threadIdx.x >> 5, lane = threadIdx.x & 31;
  int ln = lane & 15, hi = lane >> 4;
  int wn0 = (blockIdx.x * 8 + wave) * 64;
  int row0 = mt * 16;
  v8f acc[4] = {};
  for (int kc = 0; kc < Kd; kc += KCH) {
    __syncthreads();
#if HAVE_TDM
    if (threadIdx.x < 32)
      tdm_load_a_tile(A + (size_t)row0 * Kd + kc, sA, (unsigned)(Kd - kc), 16u,
                      (unsigned)KCH, 16u, (unsigned)Kd);
#else
    {
      int r = threadIdx.x >> 4, seg = threadIdx.x & 15;
      *(v8bf*)&sA[r * KCH + seg * 8] =
          *(const v8bf*)(A + (size_t)(row0 + r) * Kd + kc + seg * 8);
    }
#endif
    __syncthreads();
    if (wn0 < Nn) {
      const bf16* wbase = W + (size_t)(wn0 + ln) * Kd + kc + 8 * hi;
      if (kc + KCH < Kd) __builtin_prefetch(wbase + KCH, 0, 1);
#pragma unroll
      for (int kk = 0; kk < KCH; kk += 32) {
        v16bf a = ld_a_pair(&sA[ln * KCH + kk + 8 * hi]);
#pragma unroll
        for (int u = 0; u < 4; ++u) {
          v16bf bb = ld_b_cont(wbase + (size_t)u * 16 * Kd + kk);
          acc[u] = wmma_bf16(a, bb, acc[u]);
        }
      }
    }
  }
  if (wn0 < Nn) {
#pragma unroll
    for (int u = 0; u < 4; ++u)
#pragma unroll
      for (int j = 0; j < 8; ++j) {
        int rr = row0 + j + 8 * hi, cc = wn0 + u * 16 + ln;
        C[(size_t)rr * Nn + cc] = (bf16)(acc[u][j] + bias[cc]);
      }
  }
}

__global__ void gemm_res_f32out(const bf16* __restrict__ A, const bf16* __restrict__ W,
                                const float* __restrict__ bias, const float* __restrict__ Res,
                                float* __restrict__ C, int Nn, int Kd) {
  __shared__ bf16 sA[16 * KCH];
  int mt = blockIdx.y;
  int wave = threadIdx.x >> 5, lane = threadIdx.x & 31;
  int ln = lane & 15, hi = lane >> 4;
  int wn0 = (blockIdx.x * 8 + wave) * 64;
  int row0 = mt * 16;
  v8f acc[4] = {};
  for (int kc = 0; kc < Kd; kc += KCH) {
    __syncthreads();
#if HAVE_TDM
    if (threadIdx.x < 32)
      tdm_load_a_tile(A + (size_t)row0 * Kd + kc, sA, (unsigned)(Kd - kc), 16u,
                      (unsigned)KCH, 16u, (unsigned)Kd);
#else
    {
      int r = threadIdx.x >> 4, seg = threadIdx.x & 15;
      *(v8bf*)&sA[r * KCH + seg * 8] =
          *(const v8bf*)(A + (size_t)(row0 + r) * Kd + kc + seg * 8);
    }
#endif
    __syncthreads();
    if (wn0 < Nn) {
      const bf16* wbase = W + (size_t)(wn0 + ln) * Kd + kc + 8 * hi;
#pragma unroll
      for (int kk = 0; kk < KCH; kk += 32) {
        v16bf a = ld_a_pair(&sA[ln * KCH + kk + 8 * hi]);
#pragma unroll
        for (int u = 0; u < 4; ++u) {
          v16bf bb = ld_b_cont(wbase + (size_t)u * 16 * Kd + kk);
          acc[u] = wmma_bf16(a, bb, acc[u]);
        }
      }
    }
  }
  if (wn0 < Nn) {
#pragma unroll
    for (int u = 0; u < 4; ++u)
#pragma unroll
      for (int j = 0; j < 8; ++j) {
        int rr = row0 + j + 8 * hi, cc = wn0 + u * 16 + ln;
        C[(size_t)rr * Nn + cc] = acc[u][j] + bias[cc] + Res[(size_t)rr * Nn + cc];
      }
  }
}

// MoE GEMM1: h[slot, DF] = qgelu(xn2[token] @ wfc_e^T + bfc_e), gathered rows
__global__ void gemm_moe_fc(const bf16* __restrict__ X, const bf16* __restrict__ Wfc,
                            const float* __restrict__ Bfc, const int* __restrict__ counts,
                            const int* __restrict__ slotlist, const int* __restrict__ tokOf,
                            bf16* __restrict__ Hb) {
  int e = blockIdx.z;
  int cnt = counts[e];
  int mt = blockIdx.y;
  if (mt * 16 >= cnt) return;
  __shared__ bf16 sA[16 * KCH];
  __shared__ int sSlot[16];
  __shared__ int sTok[16];
  if (threadIdx.x < 16) {
    int p = mt * 16 + threadIdx.x;
    int s = (p < cnt) ? slotlist[e * T_ + p] : -1;
    sSlot[threadIdx.x] = s;
    sTok[threadIdx.x] = (s >= 0) ? tokOf[s] : -1;
  }
  __syncthreads();
  int wave = threadIdx.x >> 5, lane = threadIdx.x & 31;
  int ln = lane & 15, hi = lane >> 4;
  int wn0 = (blockIdx.x * 8 + wave) * 64;
  const bf16* We = Wfc + (size_t)e * DF_ * D_;
  v8f acc[4] = {};
  for (int kc = 0; kc < D_; kc += KCH) {
    __syncthreads();
    {
      int r = threadIdx.x >> 4, seg = threadIdx.x & 15;
      int t = sTok[r];
      if (t >= 0) {
        *(v8bf*)&sA[r * KCH + seg * 8] =
            *(const v8bf*)(X + (size_t)t * D_ + kc + seg * 8);
      } else {
#pragma unroll
        for (int q = 0; q < 8; ++q) sA[r * KCH + seg * 8 + q] = (bf16)0.0f;
      }
    }
    __syncthreads();
    if (wn0 < DF_) {
      const bf16* wbase = We + (size_t)(wn0 + ln) * D_ + kc + 8 * hi;
#pragma unroll
      for (int kk = 0; kk < KCH; kk += 32) {
        v16bf a = ld_a_pair(&sA[ln * KCH + kk + 8 * hi]);
#pragma unroll
        for (int u = 0; u < 4; ++u) {
          v16bf bb = ld_b_cont(wbase + (size_t)u * 16 * D_ + kk);
          acc[u] = wmma_bf16(a, bb, acc[u]);
        }
      }
    }
  }
  if (wn0 < DF_) {
#pragma unroll
    for (int u = 0; u < 4; ++u)
#pragma unroll
      for (int j = 0; j < 8; ++j) {
        int s = sSlot[j + 8 * hi];
        if (s >= 0) {
          int cc = wn0 + u * 16 + ln;
          float h = acc[u][j] + Bfc[e * DF_ + cc];
          h = h / (1.0f + __expf(-1.702f * h));  // x * sigmoid(1.702 x)
          Hb[(size_t)s * DF_ + cc] = (bf16)h;
        }
      }
  }
}

// MoE GEMM2: Y[token] += w_slot * (h[slot] @ wproj_e^T + bproj_e), scatter-add
__global__ void gemm_moe_proj(const bf16* __restrict__ Hb, const bf16* __restrict__ Wp,
                              const float* __restrict__ Bp, const int* __restrict__ counts,
                              const int* __restrict__ slotlist, const int* __restrict__ tokOf,
                              const float* __restrict__ wOf, float* __restrict__ Y) {
  int e = blockIdx.z;
  int cnt = counts[e];
  int mt = blockIdx.y;
  if (mt * 16 >= cnt) return;
  __shared__ bf16 sA[16 * KCH];
  __shared__ int sSlot[16];
  __shared__ int sTok[16];
  __shared__ float sW[16];
  if (threadIdx.x < 16) {
    int p = mt * 16 + threadIdx.x;
    int s = (p < cnt) ? slotlist[e * T_ + p] : -1;
    sSlot[threadIdx.x] = s;
    sTok[threadIdx.x] = (s >= 0) ? tokOf[s] : -1;
    sW[threadIdx.x] = (s >= 0) ? wOf[s] : 0.0f;
  }
  __syncthreads();
  int wave = threadIdx.x >> 5, lane = threadIdx.x & 31;
  int ln = lane & 15, hi = lane >> 4;
  int wn0 = (blockIdx.x * 8 + wave) * 64;
  const bf16* We = Wp + (size_t)e * D_ * DF_;
  v8f acc[4] = {};
  for (int kc = 0; kc < DF_; kc += KCH) {
    __syncthreads();
    {
      int r = threadIdx.x >> 4, seg = threadIdx.x & 15;
      int s = sSlot[r];
      if (s >= 0) {
        *(v8bf*)&sA[r * KCH + seg * 8] =
            *(const v8bf*)(Hb + (size_t)s * DF_ + kc + seg * 8);
      } else {
#pragma unroll
        for (int q = 0; q < 8; ++q) sA[r * KCH + seg * 8 + q] = (bf16)0.0f;
      }
    }
    __syncthreads();
    if (wn0 < D_) {
      const bf16* wbase = We + (size_t)(wn0 + ln) * DF_ + kc + 8 * hi;
#pragma unroll
      for (int kk = 0; kk < KCH; kk += 32) {
        v16bf a = ld_a_pair(&sA[ln * KCH + kk + 8 * hi]);
#pragma unroll
        for (int u = 0; u < 4; ++u) {
          v16bf bb = ld_b_cont(wbase + (size_t)u * 16 * DF_ + kk);
          acc[u] = wmma_bf16(a, bb, acc[u]);
        }
      }
    }
  }
  if (wn0 < D_) {
#pragma unroll
    for (int u = 0; u < 4; ++u)
#pragma unroll
      for (int j = 0; j < 8; ++j) {
        int s = sSlot[j + 8 * hi];
        if (s >= 0) {
          int t = sTok[j + 8 * hi];
          int cc = wn0 + u * 16 + ln;
          atomicAdd(&Y[(size_t)t * D_ + cc],
                    sW[j + 8 * hi] * (acc[u][j] + Bp[e * D_ + cc]));
        }
      }
  }
}

// ---------------- attention: 1 wave per (qtile, head, batch) -----------------
__global__ void attn_kernel(const bf16* __restrict__ QKV, bf16* __restrict__ AO) {
  __shared__ float sc[16 * SC_W];
  __shared__ bf16 vt[32 * HD_];  // staged V tile: 32 keys x 64 dims
  int qt = blockIdx.x, h = blockIdx.y, b = blockIdx.z;
  int lane = threadIdx.x & 31;
  int ln = lane & 15, hi = lane >> 4;

  // Q fragments: A = Q[16 x 64], two k-steps of 32
  int qn = qt * 16 + ln;
  bool qv = qn < N_;
  const bf16* qrow = QKV + ((size_t)qn * B_ + b) * QKVLD + h * HD_;
  v16bf qa[2];
#pragma unroll
  for (int s = 0; s < 2; ++s)
    qa[s] = qv ? ld_a_pair(qrow + s * 32 + 8 * hi) : zero16();

  // scores = Q @ K^T, tiled over 14 key tiles
  for (int kt = 0; kt < QT_; ++kt) {
    v8f c = {};
    int kn = kt * 16 + ln;
    bool kv = kn < N_;
    const bf16* krow = QKV + ((size_t)kn * B_ + b) * QKVLD + D_ + h * HD_;
#pragma unroll
    for (int s = 0; s < 2; ++s) {
      v16bf kb = kv ? ld_b_cont(krow + s * 32 + 8 * hi) : zero16();
      c = wmma_bf16(qa[s], kb, c);
    }
#pragma unroll
    for (int j = 0; j < 8; ++j) {
      int r = j + 8 * hi;
      int col = kt * 16 + ln;
      float v = c[j] * 0.125f;  // 1/sqrt(64)
      if (col >= N_) v = -1e9f;
      sc[r * SC_W + col] = v;
    }
  }
  __syncthreads();

  // softmax per query row (lanes 0..15 each own a row)
  if (lane < 16) {
    float* row = &sc[lane * SC_W];
    float mx = -1e30f;
    for (int i = 0; i < SC_W; ++i) mx = fmaxf(mx, row[i]);
    float sum = 0.0f;
    for (int i = 0; i < SC_W; ++i) { float ev = __expf(row[i] - mx); row[i] = ev; sum += ev; }
    float inv = 1.0f / sum;
    for (int i = 0; i < SC_W; ++i) row[i] *= inv;
  }
  __syncthreads();

  // O = P @ V : K loop of 32 keys (V tile staged in LDS), 4 column tiles (HD=64)
  v8f o[4] = {};
  for (int k0 = 0; k0 < SC_W; k0 += 32) {
    __syncthreads();
    {
      int key = k0 + lane;
      if (key < N_) {
        const bf16* vrow = QKV + ((size_t)key * B_ + b) * QKVLD + 2 * D_ + h * HD_;
#pragma unroll
        for (int q = 0; q < 8; ++q)
          *(v8bf*)&vt[lane * HD_ + q * 8] = *(const v8bf*)(vrow + q * 8);
      } else {
#pragma unroll
        for (int q = 0; q < HD_; ++q) vt[lane * HD_ + q] = (bf16)0.0f;
      }
    }
    __syncthreads();
    v16bf pa;
#pragma unroll
    for (int e = 0; e < 16; ++e) {
      int kk = k0 + 16 * hi + e;
      pa[e] = (bf16)sc[ln * SC_W + kk];
    }
#pragma unroll
    for (int nt = 0; nt < 4; ++nt) {
      int col = nt * 16 + ln;
      v16bf vb;
#pragma unroll
      for (int e = 0; e < 16; ++e)
        vb[e] = vt[(16 * hi + e) * HD_ + col];
      o[nt] = wmma_bf16(pa, vb, o[nt]);
    }
  }
#pragma unroll
  for (int nt = 0; nt < 4; ++nt) {
#pragma unroll
    for (int j = 0; j < 8; ++j) {
      int r = qt * 16 + j + 8 * hi;
      if (r < N_)
        AO[((size_t)r * B_ + b) * D_ + h * HD_ + nt * 16 + ln] = (bf16)o[nt][j];
    }
  }
}

// ---------------- gating / routing ------------------------------------------

__global__ void group_kernel(const bf16* __restrict__ Xn2, const float* __restrict__ img_w,
                             const float* __restrict__ img_b, const float* __restrict__ gumbel,
                             int* __restrict__ group_id) {
  int b = threadIdx.x;
  if (b >= B_) return;
  const bf16* xr = Xn2 + ((size_t)b * N_ + (N_ - 1)) * D_;
  float l0 = img_b[0], l1 = img_b[1];
  for (int i = 0; i < D_; ++i) {
    float xv = (float)xr[i];
    l0 += xv * img_w[i];
    l1 += xv * img_w[D_ + i];
  }
  l0 += gumbel[b * 2];
  l1 += gumbel[b * 2 + 1];
  group_id[b] = (l1 > l0) ? 1 : 0;
}

__global__ void route_kernel(const bf16* __restrict__ Xn2, const bf16* __restrict__ fea_w,
                             const float* __restrict__ fea_b, const int* __restrict__ group_id,
                             int* __restrict__ counts, int* __restrict__ slotlist,
                             int* __restrict__ tokOf, float* __restrict__ wOf) {
  int t = blockIdx.x * blockDim.x + threadIdx.x;
  if (t >= T_) return;
  int b = t / N_;
  const bf16* xr = Xn2 + (size_t)t * D_;
  float lg[E_];
#pragma unroll
  for (int e = 0; e < E_; ++e) {
    float s = fea_b[e];
    const bf16* wr = fea_w + (size_t)e * D_;
    for (int i = 0; i < D_; ++i) s += (float)xr[i] * (float)wr[i];
    lg[e] = s;
  }
  int gid = group_id[b];
  if (gid == 0) lg[E_ - 1] += -1e9f; else lg[0] += -1e9f;
  float mx = lg[0];
#pragma unroll
  for (int e = 1; e < E_; ++e) mx = fmaxf(mx, lg[e]);
  float sum = 0.0f, sm[E_];
#pragma unroll
  for (int e = 0; e < E_; ++e) { sm[e] = __expf(lg[e] - mx); sum += sm[e]; }
  float inv = 1.0f / sum;
#pragma unroll
  for (int e = 0; e < E_; ++e) sm[e] *= inv;
  int i1 = 0; float v1 = sm[0];
#pragma unroll
  for (int e = 1; e < E_; ++e) if (sm[e] > v1) { v1 = sm[e]; i1 = e; }
  int i2 = -1; float v2 = -1.0f;
#pragma unroll
  for (int e = 0; e < E_; ++e) if (e != i1 && sm[e] > v2) { v2 = sm[e]; i2 = e; }
  float norm = 1.0f / (v1 + v2 + 1e-6f);
  float w1 = v1 * norm, w2 = v2 * norm;

  int s1 = atomicAdd(&counts[8], 1);
  int p1 = atomicAdd(&counts[i1], 1);
  slotlist[i1 * T_ + p1] = s1;
  tokOf[s1] = t; wOf[s1] = w1;

  int s2 = atomicAdd(&counts[8], 1);
  int p2 = atomicAdd(&counts[i2], 1);
  slotlist[i2 * T_ + p2] = s2;
  tokOf[s2] = t; wOf[s2] = w2;
}

__global__ void final_add(const float* __restrict__ X2, const float* __restrict__ Y,
                          float* __restrict__ Out) {
  int i = blockIdx.x * blockDim.x + threadIdx.x;
  if (i >= T_ * D_) return;
  int r = i / D_, d = i % D_;
  int b = r % B_, n = r / B_;
  Out[i] = X2[i] + Y[((size_t)b * N_ + n) * D_ + d];
}

// ---------------- host side --------------------------------------------------

extern "C" void kernel_launch(void* const* d_in, const int* in_sizes, int n_in,
                              void* d_out, int out_size, void* d_ws, size_t ws_size,
                              hipStream_t stream) {
  const float* x      = (const float*)d_in[0];
  const float* gumbel = (const float*)d_in[1];
  const float* ln1_g  = (const float*)d_in[2];
  const float* ln1_b  = (const float*)d_in[3];
  const float* in_w   = (const float*)d_in[4];
  const float* in_b   = (const float*)d_in[5];
  const float* out_w  = (const float*)d_in[6];
  const float* out_b  = (const float*)d_in[7];
  const float* ln2_g  = (const float*)d_in[8];
  const float* ln2_b  = (const float*)d_in[9];
  const float* img_w  = (const float*)d_in[10];
  const float* img_b  = (const float*)d_in[11];
  const float* fea_w  = (const float*)d_in[12];
  const float* fea_b  = (const float*)d_in[13];
  const float* wfc    = (const float*)d_in[14];
  const float* bfc    = (const float*)d_in[15];
  const float* wproj  = (const float*)d_in[16];
  const float* bproj  = (const float*)d_in[17];
  float* out = (float*)d_out;
  (void)in_sizes; (void)n_in; (void)out_size; (void)ws_size;

  char* base = (char*)d_ws;
  size_t off = 0;
  auto alloc = [&](size_t bytes) -> void* {
    void* p = base + off;
    off = (off + bytes + 255) & ~(size_t)255;
    return p;
  };

  bf16*  in_w_bf  = (bf16*)alloc((size_t)3 * D_ * D_ * 2);
  bf16*  out_w_bf = (bf16*)alloc((size_t)D_ * D_ * 2);
  bf16*  fea_w_bf = (bf16*)alloc((size_t)E_ * D_ * 2);
  bf16*  wfc_bf   = (bf16*)alloc((size_t)E_ * DF_ * D_ * 2);
  bf16*  wproj_bf = (bf16*)alloc((size_t)E_ * D_ * DF_ * 2);
  bf16*  xn_bf    = (bf16*)alloc((size_t)T_ * D_ * 2);
  bf16*  qkv_bf   = (bf16*)alloc((size_t)T_ * QKVLD * 2);
  bf16*  ao_bf    = (bf16*)alloc((size_t)T_ * D_ * 2);
  float* x2       = (float*)alloc((size_t)T_ * D_ * 4);
  bf16*  xn2_bf   = (bf16*)alloc((size_t)T_ * D_ * 2);
  float* Y        = (float*)alloc((size_t)T_ * D_ * 4);
  int*   counts   = (int*)alloc(64);
  int*   group_id = (int*)alloc(B_ * 4);
  int*   slotlist = (int*)alloc((size_t)E_ * T_ * 4);
  int*   tokOf    = (int*)alloc((size_t)(2 * T_ + 64) * 4);
  float* wOf      = (float*)alloc((size_t)(2 * T_ + 64) * 4);
  bf16*  Hbuf     = (bf16*)alloc((size_t)(2 * T_ + 64) * DF_ * 2);

  // 1) weight conversions fp32 -> bf16
  {
    int n;
    n = 3 * D_ * D_;     cvt_f32_bf16<<<(n + 255) / 256, 256, 0, stream>>>(in_w, in_w_bf, n);
    n = D_ * D_;         cvt_f32_bf16<<<(n + 255) / 256, 256, 0, stream>>>(out_w, out_w_bf, n);
    n = E_ * D_;         cvt_f32_bf16<<<(n + 255) / 256, 256, 0, stream>>>(fea_w, fea_w_bf, n);
    n = E_ * DF_ * D_;   cvt_f32_bf16<<<(n + 255) / 256, 256, 0, stream>>>(wfc, wfc_bf, n);
    n = E_ * D_ * DF_;   cvt_f32_bf16<<<(n + 255) / 256, 256, 0, stream>>>(wproj, wproj_bf, n);
  }

  // 2) LN1 -> xn (bf16, [T,D], row n*B+b)
  ln_kernel<<<T_, 256, 0, stream>>>(x, ln1_g, ln1_b, xn_bf, 0);

  // 3) QKV = xn @ in_w^T + in_b  -> bf16 [T, 2304]; 36 wave-tiles of 16x64
  gemm_bias_bf16out<<<dim3((QKVLD / 64 + 7) / 8, T_ / 16), 256, 0, stream>>>(
      xn_bf, in_w_bf, in_b, qkv_bf, QKVLD, D_);

  // 4) attention -> ao (bf16)
  attn_kernel<<<dim3(QT_, H_, B_), 32, 0, stream>>>(qkv_bf, ao_bf);

  // 5) x2 = ao @ out_w^T + out_b + x (fp32 residual)
  gemm_res_f32out<<<dim3((D_ / 64 + 7) / 8, T_ / 16), 256, 0, stream>>>(
      ao_bf, out_w_bf, out_b, x, x2, D_, D_);

  // 6) LN2 -> xn2 (bf16, remapped to [B*N, D])
  ln_kernel<<<T_, 256, 0, stream>>>(x2, ln2_g, ln2_b, xn2_bf, 1);

  // 7) group selection + routing
  group_kernel<<<1, 32, 0, stream>>>(xn2_bf, img_w, img_b, gumbel, group_id);
  zero_f32<<<(T_ * D_ + 255) / 256, 256, 0, stream>>>(Y, T_ * D_);
  zero_i32<<<1, 64, 0, stream>>>(counts, 16);
  route_kernel<<<(T_ + 255) / 256, 256, 0, stream>>>(xn2_bf, fea_w_bf, fea_b, group_id,
                                                     counts, slotlist, tokOf, wOf);

  // 8) MoE expert GEMMs (routed, top-2): H = qgelu(x@wfc^T+b); Y += w*(H@wproj^T+b)
  gemm_moe_fc<<<dim3(DF_ / 64 / 8, T_ / 16, E_), 256, 0, stream>>>(
      xn2_bf, wfc_bf, bfc, counts, slotlist, tokOf, Hbuf);
  gemm_moe_proj<<<dim3((D_ / 64 + 7) / 8, T_ / 16, E_), 256, 0, stream>>>(
      Hbuf, wproj_bf, bproj, counts, slotlist, tokOf, wOf, Y);

  // 9) out = x2 + y (remap [B,N,D] -> [N,B,D])
  final_add<<<(T_ * D_ + 255) / 256, 256, 0, stream>>>(x2, Y, out);
}
